// ProbSparseAttention_14594298872399
// MI455X (gfx1250) — matrix operand order repaired and checked by reading, not compiled, not run on hardware
//
#include <hip/hip_runtime.h>
#include <cstdint>
#include <cstddef>

#define B_  8
#define L_  4096
#define DM  512
#define H_  8
#define DH  64
#define U_  45

typedef __attribute__((ext_vector_type(16))) __bf16 v16bf;
typedef __attribute__((ext_vector_type(8)))  float  v8f;
typedef __attribute__((ext_vector_type(4)))  unsigned int v4u;
typedef __attribute__((ext_vector_type(8)))  int v8i_;
typedef __attribute__((ext_vector_type(4)))  int v4i_;

__device__ __forceinline__ unsigned short f2bf(float f) {
  unsigned u = __float_as_uint(f);
  u += 0x7FFFu + ((u >> 16) & 1u);          // round-to-nearest-even
  return (unsigned short)(u >> 16);
}

union FragB16 { v16bf v; uint32_t u[8]; uint4 q[2]; };

// ---------------------------------------------------------------------------
// Weight transpose + fp32 -> bf16:  WT[n*512 + k] = bf16(W[k*512 + n])
// ---------------------------------------------------------------------------
__global__ void wt_transpose_bf16(const float* __restrict__ W,
                                  unsigned short* __restrict__ WT) {
  int i = blockIdx.x * blockDim.x + threadIdx.x;   // 0 .. 512*512-1
  int k = i >> 9, n = i & 511;
  WT[n * DM + k] = f2bf(W[i]);
}

// ---------------------------------------------------------------------------
// C[M x 512] = A[M x 512](fp32) @ W (via WT[n][k] bf16) + bias, f32 accum.
// Block: 256 threads (8 wave32), tile 128x128; wave tile 32x64 (2x4 WMMA).
// B panel (128 n-rows x 512 k, bf16 = 128KB) DMA'd once per block into LDS
// by the Tensor Data Mover; A tile (128x32) staged fp32->bf16 per k-step.
// Dynamic LDS: 128*512*2 + 128*32*2 = 139264 bytes.
// ---------------------------------------------------------------------------
__global__ __launch_bounds__(256) void gemm_bf16_wmma(
    const float* __restrict__ A,
    const unsigned short* __restrict__ BT,   // [512 x 512] bf16, row n holds W[:,n]
    const float* __restrict__ bias,
    float* __restrict__ C,
    int M)
{
  extern __shared__ char smem[];
  unsigned short* Bpanel = (unsigned short*)smem;                    // 128 x 512
  unsigned short* Alds   = (unsigned short*)(smem + 128 * 512 * 2);  // 128 x 32

  const int t     = threadIdx.x;
  const int lane  = t & 31;
  const int wave  = t >> 5;
  const int wm    = wave >> 1;          // 0..3
  const int wn    = wave & 1;           // 0..1
  const int khalf = lane >> 4;          // 0/1: which K half-wave
  const int l16   = lane & 15;

  const int m0 = blockIdx.y * 128;
  const int n0 = blockIdx.x * 128;

  // ---- TDM: DMA the whole B panel (rows n0..n0+127, all 512 k) into LDS ----
  if (wave == 0) {
    const unsigned long long gaddr =
        (unsigned long long)(uintptr_t)(BT + (size_t)n0 * DM);
    const unsigned lds_off = (unsigned)(uintptr_t)Bpanel;
    // D# group0: count=1 | lds_addr | global_addr[56:0] | type=2 (bits 127:126)
    v4u g0 = { 1u,
               lds_off,
               (unsigned)(gaddr & 0xFFFFFFFFu),
               (unsigned)((gaddr >> 32) & 0x01FFFFFFu) | 0x80000000u };
    // D# group1: wg_mask=0, data_size=1(2B) | tensor_dim0=512 | tensor_dim1=512
    //            tile_dim0=512 | tile_dim1=128, tile_dim2=0 | dim0_stride=512
    v8i_ g1 = { 0x00010000,          // data_size=1 (2 bytes)
                (int)(512u << 16),   // tensor_dim0[15:0] in bits 63:48
                (int)(512u << 16),   // tensor_dim1[15:0] in bits 95:80
                (int)(512u << 16),   // tile_dim0=512 in bits 127:112
                128,                 // tile_dim1=128, tile_dim2=0
                512,                 // tensor_dim0_stride low 32
                0, 0 };
    v4i_ gz4 = { 0, 0, 0, 0 };           // groups 2/3: 2D tensor, unused
    v8i_ gz8 = { 0, 0, 0, 0, 0, 0, 0, 0 };
    __builtin_amdgcn_tensor_load_to_lds(g0, g1, gz4, gz4, gz8, 0);
    __builtin_amdgcn_s_wait_tensorcnt(0);
  }

  v8f acc[2][4] = {};                   // zero-initialized accumulators

  const int arow  = t >> 1;             // 0..127 : staging row
  const int ahalf = (t & 1) * 16;       // staging K sub-offset

  for (int kt = 0; kt < DM; kt += 32) {
    // --- stage A tile (128x32), fp32 -> bf16 ---
    {
      const float* src = A + (size_t)(m0 + arow) * DM + kt + ahalf;
      unsigned short* dst = Alds + arow * 32 + ahalf;
      #pragma unroll
      for (int i = 0; i < 16; i += 4) {
        float4 f = *reinterpret_cast<const float4*>(src + i);
        dst[i + 0] = f2bf(f.x); dst[i + 1] = f2bf(f.y);
        dst[i + 2] = f2bf(f.z); dst[i + 3] = f2bf(f.w);
      }
      if (kt + 32 < DM)
        __builtin_prefetch(src + 32, 0, 1);   // global_prefetch next k-slice
    }
    __syncthreads();   // A tile ready (first iter: B panel published too)

    // --- build fragments per ISA 7.12.2 16-bit layouts ---
    FragB16 afrag[2], bfrag[4];
    #pragma unroll
    for (int sm = 0; sm < 2; ++sm) {
      const int mrow = wm * 32 + sm * 16 + l16;
      const unsigned short* ap = Alds + mrow * 32;
      #pragma unroll
      for (int v = 0; v < 8; ++v) {
        const int k = (v < 4 ? 2 * v : 16 + 2 * (v - 4)) + khalf * 8;
        afrag[sm].u[v] = *reinterpret_cast<const uint32_t*>(ap + k);
      }
    }
    #pragma unroll
    for (int sn = 0; sn < 4; ++sn) {
      const int nrow = wn * 64 + sn * 16 + l16;
      const unsigned short* bp = Bpanel + nrow * DM + kt + khalf * 8;
      bfrag[sn].q[0] = *reinterpret_cast<const uint4*>(bp);
      bfrag[sn].q[1] = *reinterpret_cast<const uint4*>(bp + 16);
    }

    #pragma unroll
    for (int sm = 0; sm < 2; ++sm)
      #pragma unroll
      for (int sn = 0; sn < 4; ++sn)
        acc[sm][sn] = __builtin_amdgcn_wmma_f32_16x16x32_bf16(
            false, afrag[sm].v, false, bfrag[sn].v,
            (short)0, acc[sm][sn], false, false);

    __syncthreads();   // protect Alds before next staging
  }

  // --- epilogue: C/D layout lane=(n), vgpr r => m = r + 8*khalf ---
  #pragma unroll
  for (int sm = 0; sm < 2; ++sm) {
    #pragma unroll
    for (int sn = 0; sn < 4; ++sn) {
      const int col   = n0 + wn * 64 + sn * 16 + l16;
      const float bc  = bias[col];
      const int rbase = m0 + wm * 32 + sm * 16 + khalf * 8;
      #pragma unroll
      for (int r = 0; r < 8; ++r)
        C[(size_t)(rbase + r) * DM + col] = acc[sm][sn][r] + bc;
    }
  }
}

// ---------------------------------------------------------------------------
// M[b,h,q] = max_s(scale*Q·K_sample) - mean_s(scale*Q·K_sample)
// ---------------------------------------------------------------------------
__global__ __launch_bounds__(256) void sampled_score_kernel(
    const float* __restrict__ Q, const float* __restrict__ K,
    const int* __restrict__ sample_idx, float* __restrict__ Msc)
{
  __shared__ float Ks[U_ * DH];
  const int bh = blockIdx.x;
  const int b = bh >> 3, h = bh & 7;
  const int q = blockIdx.y * 256 + threadIdx.x;

  for (int i = threadIdx.x; i < U_ * DH; i += 256) {
    const int s = i >> 6, d = i & 63;
    Ks[i] = K[(size_t)(b * L_ + sample_idx[s]) * DM + h * DH + d];
  }
  __syncthreads();

  float qr[DH];
  const float* qp = Q + (size_t)(b * L_ + q) * DM + h * DH;
  #pragma unroll
  for (int d = 0; d < DH; d += 4) {
    float4 f = *reinterpret_cast<const float4*>(qp + d);
    qr[d] = f.x; qr[d + 1] = f.y; qr[d + 2] = f.z; qr[d + 3] = f.w;
  }

  float mx = -3.4e38f, sum = 0.f;
  for (int s = 0; s < U_; ++s) {
    const float* kp = Ks + s * DH;
    float dp = 0.f;
    #pragma unroll
    for (int d = 0; d < DH; ++d) dp += qr[d] * kp[d];
    dp *= 0.125f;
    mx = fmaxf(mx, dp);
    sum += dp;
  }
  Msc[(size_t)bh * L_ + q] = mx - sum * (1.0f / U_);
}

// ---------------------------------------------------------------------------
// Iterative top-45 per (b,h) over L=4096
// ---------------------------------------------------------------------------
__global__ __launch_bounds__(256) void topk_kernel(const float* __restrict__ Msc,
                                                   int* __restrict__ topIdx)
{
  __shared__ float mv[L_];
  __shared__ float rv[256];
  __shared__ int   ri[256];
  const int bh = blockIdx.x;
  const int t  = threadIdx.x;

  for (int i = t; i < L_; i += 256) mv[i] = Msc[(size_t)bh * L_ + i];
  __syncthreads();

  for (int it = 0; it < U_; ++it) {
    float best = -3.4e38f; int bi = 0;
    for (int i = t; i < L_; i += 256) {
      const float v = mv[i];
      if (v > best) { best = v; bi = i; }
    }
    rv[t] = best; ri[t] = bi;
    __syncthreads();
    for (int s = 128; s > 0; s >>= 1) {
      if (t < s && rv[t + s] > rv[t]) { rv[t] = rv[t + s]; ri[t] = ri[t + s]; }
      __syncthreads();
    }
    if (t == 0) { topIdx[bh * U_ + it] = ri[0]; mv[ri[0]] = -3.4e38f; }
    __syncthreads();
  }
}

// ---------------------------------------------------------------------------
// Full softmax attention for the 45 selected queries per (b,h)
// ---------------------------------------------------------------------------
__global__ __launch_bounds__(256) void topq_attention_kernel(
    const float* __restrict__ Q, const float* __restrict__ K,
    const float* __restrict__ V, const int* __restrict__ topIdx,
    float* __restrict__ attnOut)
{
  __shared__ float sc[L_];
  __shared__ float qrow[DH];
  __shared__ float red[256];
  __shared__ float part[4][DH];

  const int j  = blockIdx.x;              // 0..B*H*U-1
  const int bh = j / U_;
  const int b  = bh >> 3, h = bh & 7;
  const int t  = threadIdx.x;
  const int qi = topIdx[j];

  if (t < DH) qrow[t] = Q[(size_t)(b * L_ + qi) * DM + h * DH + t];
  __syncthreads();

  float qreg[DH];
  #pragma unroll
  for (int d = 0; d < DH; ++d) qreg[d] = qrow[d];

  float lmax = -3.4e38f;
  for (int k = t; k < L_; k += 256) {
    const float* kp = K + (size_t)(b * L_ + k) * DM + h * DH;
    float dp = 0.f;
    #pragma unroll
    for (int d = 0; d < DH; ++d) dp += qreg[d] * kp[d];
    dp *= 0.125f;
    sc[k] = dp;
    lmax = fmaxf(lmax, dp);
  }
  red[t] = lmax; __syncthreads();
  for (int s = 128; s > 0; s >>= 1) {
    if (t < s) red[t] = fmaxf(red[t], red[t + s]);
    __syncthreads();
  }
  const float gmax = red[0];
  __syncthreads();

  float lsum = 0.f;
  for (int k = t; k < L_; k += 256) {
    const float e = __expf(sc[k] - gmax);
    sc[k] = e;
    lsum += e;
  }
  red[t] = lsum; __syncthreads();
  for (int s = 128; s > 0; s >>= 1) {
    if (t < s) red[t] += red[t + s];
    __syncthreads();
  }
  const float inv = 1.0f / red[0];

  const int d = t & 63, g = t >> 6;
  float acc = 0.f;
  const int k0 = g * (L_ / 4), k1 = k0 + (L_ / 4);
  for (int k = k0; k < k1; ++k)
    acc += sc[k] * V[(size_t)(b * L_ + k) * DM + h * DH + d];
  part[g][d] = acc;
  __syncthreads();

  if (t < DH) {
    const float s = (part[0][t] + part[1][t]) + (part[2][t] + part[3][t]);
    attnOut[(size_t)j * DH + t] = s * inv;
  }
}

// ---------------------------------------------------------------------------
// V mean over L per (b,h,d)
// ---------------------------------------------------------------------------
__global__ __launch_bounds__(256) void vmean_kernel(const float* __restrict__ V,
                                                    float* __restrict__ vmean)
{
  __shared__ float red[256];
  const int bh = blockIdx.x;
  const int b = bh >> 3, h = bh & 7;
  const int t = threadIdx.x;
  const int d = t & 63, g = t >> 6;
  float acc = 0.f;
  for (int l = g * (L_ / 4); l < (g + 1) * (L_ / 4); ++l)
    acc += V[(size_t)(b * L_ + l) * DM + h * DH + d];
  red[t] = acc;
  __syncthreads();
  if (t < DH)
    vmean[bh * DH + t] =
        (red[t] + red[t + 64] + red[t + 128] + red[t + 192]) * (1.0f / L_);
}

// ---------------------------------------------------------------------------
// ctx = broadcast(vmean) over (b, l, h, d)
// ---------------------------------------------------------------------------
__global__ void ctx_broadcast_kernel(const float* __restrict__ vmean,
                                     float* __restrict__ ctx)
{
  const size_t i = (size_t)blockIdx.x * blockDim.x + threadIdx.x; // B*L*DM
  const int c   = (int)(i & 511);
  const size_t row = i >> 9;                  // b*L + l
  const int b = (int)(row >> 12);             // L = 4096
  const int h = c >> 6, d = c & 63;
  ctx[i] = vmean[((b << 3) + h) * DH + d];
}

// ---------------------------------------------------------------------------
// scatter attn_out into ctx at top query positions
// ---------------------------------------------------------------------------
__global__ void ctx_scatter_kernel(const float* __restrict__ attnOut,
                                   const int* __restrict__ topIdx,
                                   float* __restrict__ ctx)
{
  const int i = blockIdx.x * blockDim.x + threadIdx.x;
  if (i >= B_ * H_ * U_ * DH) return;
  const int j = i >> 6, d = i & 63;
  const int bh = j / U_;
  const int b = bh >> 3, h = bh & 7;
  const int l = topIdx[j];
  ctx[(size_t)(b * L_ + l) * DM + h * DH + d] = attnOut[i];
}

// ---------------------------------------------------------------------------
extern "C" void kernel_launch(void* const* d_in, const int* in_sizes, int n_in,
                              void* d_out, int out_size, void* d_ws, size_t ws_size,
                              hipStream_t stream)
{
  (void)in_sizes; (void)n_in; (void)out_size; (void)ws_size;

  const float* queries = (const float*)d_in[0];
  const float* keys    = (const float*)d_in[1];
  const float* values  = (const float*)d_in[2];
  const int*   sample  = (const int*)d_in[3];
  const float* Wq = (const float*)d_in[4];  const float* bq = (const float*)d_in[5];
  const float* Wk = (const float*)d_in[6];  const float* bk = (const float*)d_in[7];
  const float* Wv = (const float*)d_in[8];  const float* bv = (const float*)d_in[9];
  const float* Wo = (const float*)d_in[10]; const float* bo = (const float*)d_in[11];
  float* out = (float*)d_out;

  char* ws = (char*)d_ws;
  size_t cur = 0;
  auto alloc = [&](size_t bytes) -> void* {
    cur = (cur + 255) & ~(size_t)255;
    void* p = ws + cur;
    cur += bytes;
    return p;
  };

  unsigned short* WqT = (unsigned short*)alloc((size_t)DM * DM * 2);
  unsigned short* WkT = (unsigned short*)alloc((size_t)DM * DM * 2);
  unsigned short* WvT = (unsigned short*)alloc((size_t)DM * DM * 2);
  unsigned short* WoT = (unsigned short*)alloc((size_t)DM * DM * 2);
  float* Qp      = (float*)alloc((size_t)B_ * L_ * DM * 4);
  float* Kp      = (float*)alloc((size_t)B_ * L_ * DM * 4);
  float* Vp      = (float*)alloc((size_t)B_ * L_ * DM * 4);
  float* Msc     = (float*)alloc((size_t)B_ * H_ * L_ * 4);
  int*   topIdx  = (int*)alloc((size_t)B_ * H_ * U_ * 4);
  float* attnOut = (float*)alloc((size_t)B_ * H_ * U_ * DH * 4);
  float* vmean   = (float*)alloc((size_t)B_ * H_ * DH * 4);
  float* ctx     = (float*)alloc((size_t)B_ * L_ * DM * 4);

  const dim3 blk(256);
  const int M = B_ * L_;
  const size_t gemm_lds = 128 * 512 * 2 + 128 * 32 * 2;   // 139264 B

  // weight transposes (fp32 -> bf16)
  wt_transpose_bf16<<<(DM * DM) / 256, blk, 0, stream>>>(Wq, WqT);
  wt_transpose_bf16<<<(DM * DM) / 256, blk, 0, stream>>>(Wk, WkT);
  wt_transpose_bf16<<<(DM * DM) / 256, blk, 0, stream>>>(Wv, WvT);
  wt_transpose_bf16<<<(DM * DM) / 256, blk, 0, stream>>>(Wo, WoT);

  // projections (WMMA bf16, f32 accumulate; TDM-loaded weight panels)
  const dim3 ggrid(DM / 128, M / 128);
  gemm_bf16_wmma<<<ggrid, blk, gemm_lds, stream>>>(queries, WqT, bq, Qp, M);
  gemm_bf16_wmma<<<ggrid, blk, gemm_lds, stream>>>(keys,    WkT, bk, Kp, M);
  gemm_bf16_wmma<<<ggrid, blk, gemm_lds, stream>>>(values,  WvT, bv, Vp, M);

  // probsparse measure + top-u selection
  sampled_score_kernel<<<dim3(B_ * H_, L_ / 256), blk, 0, stream>>>(Qp, Kp, sample, Msc);
  topk_kernel<<<B_ * H_, blk, 0, stream>>>(Msc, topIdx);

  // attention for selected queries; mean-V context elsewhere
  topq_attention_kernel<<<B_ * H_ * U_, blk, 0, stream>>>(Qp, Kp, Vp, topIdx, attnOut);
  vmean_kernel<<<B_ * H_, blk, 0, stream>>>(Vp, vmean);
  ctx_broadcast_kernel<<<(int)(((size_t)B_ * L_ * DM) / 256), blk, 0, stream>>>(vmean, ctx);
  ctx_scatter_kernel<<<(B_ * H_ * U_ * DH + 255) / 256, blk, 0, stream>>>(attnOut, topIdx, ctx);

  // output projection (WMMA bf16) -> d_out
  gemm_bf16_wmma<<<ggrid, blk, gemm_lds, stream>>>(ctx, WoT, bo, out, M);
}
